// BioSNN_4913442587056
// MI455X (gfx1250) — matrix operand order repaired and checked by reading, not compiled, not run on hardware
//
#include <hip/hip_runtime.h>

// ---------------------------------------------------------------------------
// Types for CDNA5 WMMA (wave32)
// ---------------------------------------------------------------------------
typedef __bf16 v16bf __attribute__((ext_vector_type(16)));
typedef __bf16 v8bf  __attribute__((ext_vector_type(8)));
typedef float  v8f   __attribute__((ext_vector_type(8)));

union FragU { v16bf v; v8bf h[2]; };

// A-matrix fragment, 16x32 bf16 (ISA 7.12.2 interleaved-K layout).
__device__ __forceinline__ v16bf load_frag_a(const __bf16* __restrict__ base,
                                             int ld, int k, int lane) {
  const int r  = lane & 15;
  const int ko = (lane >> 4) << 3;            // 0 or 8
  const __bf16* p = base + (size_t)r * ld + (size_t)(k + ko);
  FragU u;
  u.h[0] = *(const v8bf*)(p);                 // K = k+ko    .. k+ko+7
  u.h[1] = *(const v8bf*)(p + 16);            // K = k+ko+16 .. k+ko+23
  return u.v;
}

// B-matrix fragment, 32x16 bf16 (K x N), source W row-major [N][ld] (B = W^T).
__device__ __forceinline__ v16bf load_frag_b(const __bf16* __restrict__ base,
                                             int ld, int k, int lane) {
  const int r  = lane & 15;
  const int ko = (lane >> 4) << 3;            // 0 or 8
  const __bf16* p = base + (size_t)r * ld + (size_t)(k + ko);
  FragU u;
  u.h[0] = *(const v8bf*)(p);
  u.h[1] = *(const v8bf*)(p + 16);
  return u.v;
}

#define WMMA_BF16(a, b, c) \
  __builtin_amdgcn_wmma_f32_16x16x32_bf16(false, (a), false, (b), (short)0, (c), false, false)

__device__ __forceinline__ void wait_asynccnt0() {
#if __has_builtin(__builtin_amdgcn_s_wait_asynccnt)
  __builtin_amdgcn_s_wait_asynccnt(0);
#else
  asm volatile("s_wait_asynccnt 0x0" ::: "memory");
#endif
}

// ---------------------------------------------------------------------------
// f32 -> bf16 conversion with zero padding (rows_p x cols_p destination)
// ---------------------------------------------------------------------------
__global__ void __launch_bounds__(256)
k_cvt_pad(const float* __restrict__ src, __bf16* __restrict__ dst,
          int rows, int cols, int rows_p, int cols_p) {
  size_t gid = (size_t)blockIdx.x * blockDim.x + threadIdx.x;
  size_t total = (size_t)rows_p * cols_p;
  if (gid >= total) return;
  int r = (int)(gid / cols_p);
  int c = (int)(gid % cols_p);
  float v = (r < rows && c < cols) ? src[(size_t)r * cols + c] : 0.0f;
  dst[gid] = (__bf16)v;
}

// ---------------------------------------------------------------------------
// LDS-staged WMMA GEMM for N == 800, Kp == 800:
//   C[M x 800](f32) = A[M x 800](bf16) @ B[800 x 800]^T
// One block = 25 waves (800 threads) = one 64(M) x 800(N) stripe.
// The shared 64x800 A panel is staged ONCE into LDS with async global->LDS
// copies (tracked by ASYNCcnt), then every wave feeds its A fragments from
// LDS (ds_load_b128). Row stride padded to 808 bf16 for bank-conflict-free
// 16-row b128 reads. B fragments come straight from global (W is L2-resident).
// ---------------------------------------------------------------------------
#define KP800 800
#define LDSW  808   // padded LDS row stride in bf16 elements (1616 B)

__global__ void __launch_bounds__(800, 1)
k_gemm_lds(const __bf16* __restrict__ A, const __bf16* __restrict__ B,
           float* __restrict__ C) {
  extern __shared__ __align__(16) __bf16 smem[];   // 64 * LDSW bf16 = 103,424 B

  const int tid  = (int)threadIdx.x;               // 0..799
  const int wave = tid >> 5;                       // 0..24  -> N tile
  const int lane = tid & 31;
  const size_t m0 = (size_t)blockIdx.x * 64;
  const size_t n0 = (size_t)wave * 32;

  // ---- stage A panel: 64 rows x 800 cols = 6400 x 16B chunks, async to LDS
  const uint32_t lds_base = (uint32_t)(uintptr_t)smem;  // LDS offset = addr[31:0]
#pragma unroll
  for (int i = 0; i < 8; ++i) {
    const int c   = tid + i * 800;                 // chunk id 0..6399
    const int row = c / 100;                       // 0..63
    const int col = c - row * 100;                 // 16B chunk within row
    const uint32_t ldso = lds_base + (uint32_t)(row * (LDSW * 2) + col * 16);
    const uint64_t gaddr =
        (uint64_t)(uintptr_t)(A + (m0 + (size_t)row) * KP800 + (size_t)col * 8);
    asm volatile("global_load_async_to_lds_b128 %0, %1, off"
                 :: "v"(ldso), "v"(gaddr) : "memory");
  }
  wait_asynccnt0();
  __syncthreads();

  const __bf16* Bp0 = B + n0 * KP800;
  const __bf16* Bp1 = Bp0 + 16 * KP800;

  v8f acc[4][2];
#pragma unroll
  for (int i = 0; i < 4; ++i)
#pragma unroll
    for (int j = 0; j < 2; ++j) acc[i][j] = (v8f){0.f,0.f,0.f,0.f,0.f,0.f,0.f,0.f};

  const int r  = lane & 15;
  const int ko = (lane >> 4) << 3;                 // A frag K sub-offset (0/8)

  for (int k = 0; k < KP800; k += 32) {
    __builtin_prefetch(Bp0 + (size_t)r * KP800 + (size_t)(k + 32), 0, 1);

    v16bf bf0 = load_frag_b(Bp0, KP800, k, lane);
    v16bf bf1 = load_frag_b(Bp1, KP800, k, lane);

    v16bf afr[4];
#pragma unroll
    for (int i = 0; i < 4; ++i) {
      const __bf16* p = smem + (size_t)(16 * i + r) * LDSW + (size_t)(k + ko);
      FragU u;
      u.h[0] = *(const v8bf*)(p);                  // ds_load_b128
      u.h[1] = *(const v8bf*)(p + 16);             // ds_load_b128
      afr[i] = u.v;
    }

#pragma unroll
    for (int i = 0; i < 4; ++i) {
      acc[i][0] = WMMA_BF16(afr[i], bf0, acc[i][0]);
      acc[i][1] = WMMA_BF16(afr[i], bf1, acc[i][1]);
    }
  }

  // C/D layout: VGPR r on lane L holds C[M = r + 8*(L>>4)][N = L&15]
  const int n  = lane & 15;
  const int mb = (lane >> 4) << 3;
#pragma unroll
  for (int i = 0; i < 4; ++i) {
    float* Crow = C + (m0 + (size_t)(16 * i) + mb) * (size_t)KP800 + n0 + n;
#pragma unroll
    for (int rr = 0; rr < 8; ++rr) {
      Crow[(size_t)rr * KP800]      = acc[i][0][rr];
      Crow[(size_t)rr * KP800 + 16] = acc[i][1][rr];
    }
  }
}

// ---------------------------------------------------------------------------
// Direct WMMA GEMM (used for the small layer-3 GEMM, N padded to 32)
// One wave computes a 64(M) x 32(N) tile.
// ---------------------------------------------------------------------------
__global__ void __launch_bounds__(256)
k_gemm_bf16(const __bf16* __restrict__ A, const __bf16* __restrict__ B,
            float* __restrict__ C, int Kp, int ldc,
            int nTilesN, int totalTiles) {
  const int wave = (int)((blockIdx.x * (size_t)blockDim.x + threadIdx.x) >> 5);
  const int lane = (int)(threadIdx.x & 31);
  if (wave >= totalTiles) return;

  const int tm = wave / nTilesN;
  const int tn = wave - tm * nTilesN;
  const size_t m0 = (size_t)tm * 64;
  const size_t n0 = (size_t)tn * 32;

  const __bf16* Ap = A + m0 * (size_t)Kp;
  const __bf16* Bp = B + n0 * (size_t)Kp;

  v8f acc[4][2];
#pragma unroll
  for (int i = 0; i < 4; ++i)
#pragma unroll
    for (int j = 0; j < 2; ++j) acc[i][j] = (v8f){0.f,0.f,0.f,0.f,0.f,0.f,0.f,0.f};

  for (int k = 0; k < Kp; k += 32) {
    __builtin_prefetch(Ap + (size_t)(lane & 15) * Kp + (size_t)(k + 32), 0, 1);

    v16bf bf0 = load_frag_b(Bp,                   Kp, k, lane);
    v16bf bf1 = load_frag_b(Bp + 16 * (size_t)Kp, Kp, k, lane);
    v16bf a0  = load_frag_a(Ap,                   Kp, k, lane);
    v16bf a1  = load_frag_a(Ap + 16 * (size_t)Kp, Kp, k, lane);
    v16bf a2  = load_frag_a(Ap + 32 * (size_t)Kp, Kp, k, lane);
    v16bf a3  = load_frag_a(Ap + 48 * (size_t)Kp, Kp, k, lane);

    acc[0][0] = WMMA_BF16(a0, bf0, acc[0][0]);
    acc[0][1] = WMMA_BF16(a0, bf1, acc[0][1]);
    acc[1][0] = WMMA_BF16(a1, bf0, acc[1][0]);
    acc[1][1] = WMMA_BF16(a1, bf1, acc[1][1]);
    acc[2][0] = WMMA_BF16(a2, bf0, acc[2][0]);
    acc[2][1] = WMMA_BF16(a2, bf1, acc[2][1]);
    acc[3][0] = WMMA_BF16(a3, bf0, acc[3][0]);
    acc[3][1] = WMMA_BF16(a3, bf1, acc[3][1]);
  }

  const int n  = lane & 15;
  const int mb = (lane >> 4) << 3;
#pragma unroll
  for (int i = 0; i < 4; ++i) {
    float* Crow = C + (m0 + (size_t)(16 * i) + mb) * (size_t)ldc + n0 + n;
#pragma unroll
    for (int rr = 0; rr < 8; ++rr) {
      Crow[(size_t)rr * ldc]      = acc[i][0][rr];
      Crow[(size_t)rr * ldc + 16] = acc[i][1][rr];
    }
  }
}

// ---------------------------------------------------------------------------
// LIF scan over T (snntorch Leaky, reset_mechanism='subtract', THR=1)
// one thread per (b,h); current = (gemm + bias[h]) * vmod[h]
// ---------------------------------------------------------------------------
__global__ void __launch_bounds__(256)
k_lif(const float* __restrict__ cur, int T, int B, int H, int ldc,
      const float* __restrict__ bias, const float* __restrict__ vmod,
      const float* __restrict__ beta_p,
      float* __restrict__ spk_out,           // [T][B][H] f32 (dense H)
      __bf16* __restrict__ spk_bf, int ldbf, // [T*B][ldbf] bf16 or null
      float* __restrict__ sum_out) {         // [B][H] or null
  const int gid = (int)(blockIdx.x * (size_t)blockDim.x + threadIdx.x);
  if (gid >= B * H) return;
  const int b = gid / H;
  const int h = gid - b * H;

  const float beta = fminf(fmaxf(beta_p[0], 0.0f), 1.0f);
  const float bia  = bias ? bias[h] : 0.0f;
  const float vm   = vmod ? vmod[h] : 1.0f;

  float mem = 0.0f, s = 0.0f;
  for (int t = 0; t < T; ++t) {
    const size_t m = (size_t)t * B + b;
    const float c = (cur[m * (size_t)ldc + h] + bia) * vm;
    const float reset = (mem > 1.0f) ? 1.0f : 0.0f;   // (mem - THR > 0)
    mem = beta * mem + c - reset;                      // reset * THR, THR=1
    const float spk = (mem > 1.0f) ? 1.0f : 0.0f;      // (mem_new - THR > 0)
    spk_out[m * (size_t)H + h] = spk;
    if (spk_bf) spk_bf[m * (size_t)ldbf + h] = (__bf16)spk;
    s += spk;
  }
  if (sum_out) sum_out[gid] = s;
}

// ---------------------------------------------------------------------------
// Launcher
// ---------------------------------------------------------------------------
extern "C" void kernel_launch(void* const* d_in, const int* in_sizes, int n_in,
                              void* d_out, int out_size, void* d_ws, size_t ws_size,
                              hipStream_t stream) {
  (void)in_sizes; (void)n_in; (void)out_size; (void)ws_size;

  const float* x   = (const float*)d_in[0];   // [25][1024][784]
  const float* W1  = (const float*)d_in[1];   // [800][784]
  const float* b1  = (const float*)d_in[2];   // [800]
  const float* W2  = (const float*)d_in[3];   // [800][800]
  const float* b2  = (const float*)d_in[4];   // [800]
  const float* W3  = (const float*)d_in[5];   // [10][800]
  const float* b3  = (const float*)d_in[6];   // [10]
  const float* v1  = (const float*)d_in[7];   // [800]
  const float* v2  = (const float*)d_in[8];   // [800]
  const float* be1 = (const float*)d_in[9];   // [1]
  const float* be2 = (const float*)d_in[10];  // [1]
  const float* be3 = (const float*)d_in[11];  // [1]

  constexpr int T = 25, Bt = 1024, IN = 784, H = 800, OUT = 10;
  constexpr int KP = 800;     // K padded to multiple of 32
  constexpr int NP3 = 32;     // layer-3 N padded (10 -> 32)
  constexpr size_t M = (size_t)T * Bt;  // 25600 rows

  constexpr size_t szXbf  = M * KP * 2;
  constexpr size_t szW1bf = (size_t)H * KP * 2;
  constexpr size_t szW2bf = (size_t)H * KP * 2;
  constexpr size_t szW3bf = (size_t)NP3 * KP * 2;
  constexpr size_t szC    = M * H * 4;          // reused for all layers
  constexpr size_t szSbf  = M * KP * 2;

  char* ws = (char*)d_ws;
  size_t off = 0;
  auto carve = [&](size_t bytes) -> char* {
    char* p = ws + off;
    off += (bytes + 255) & ~(size_t)255;
    return p;
  };
  __bf16* Xbf  = (__bf16*)carve(szXbf);
  __bf16* W1bf = (__bf16*)carve(szW1bf);
  __bf16* W2bf = (__bf16*)carve(szW2bf);
  __bf16* W3bf = (__bf16*)carve(szW3bf);
  float*  Cbuf = (float*)carve(szC);
  __bf16* S1bf = (__bf16*)carve(szSbf);
  __bf16* S2bf = (__bf16*)carve(szSbf);

  float* out   = (float*)d_out;
  float* s3sum = out;                                   // [1024][10]
  float* s1r   = out + (size_t)Bt * OUT;                // [25][1024][800]
  float* s2r   = s1r + M * H;                           // [25][1024][800]
  float* s3r   = s2r + M * H;                           // [25][1024][10]

  auto blocks1d = [](size_t n) { return dim3((unsigned)((n + 255) / 256)); };

  // 1) convert + pad to bf16
  k_cvt_pad<<<blocks1d(M * KP), 256, 0, stream>>>(x, Xbf, (int)M, IN, (int)M, KP);
  k_cvt_pad<<<blocks1d((size_t)H * KP), 256, 0, stream>>>(W1, W1bf, H, IN, H, KP);
  k_cvt_pad<<<blocks1d((size_t)H * KP), 256, 0, stream>>>(W2, W2bf, H, H, H, KP);
  k_cvt_pad<<<blocks1d((size_t)NP3 * KP), 256, 0, stream>>>(W3, W3bf, OUT, H, NP3, KP);

  const unsigned gemmBlocks = (unsigned)(M / 64);       // 400 stripes
  const size_t   ldsBytes   = (size_t)64 * LDSW * 2;    // 103,424 B

  // 2) layer 1: cur1 = Xbf @ W1^T  (M=25600, N=800, K=800), async-LDS staged
  k_gemm_lds<<<dim3(gemmBlocks), 800, ldsBytes, stream>>>(Xbf, W1bf, Cbuf);
  k_lif<<<blocks1d((size_t)Bt * H), 256, 0, stream>>>(
      Cbuf, T, Bt, H, H, b1, v1, be1, s1r, S1bf, KP, nullptr);

  // 3) layer 2: cur2 = spk1 @ W2^T
  k_gemm_lds<<<dim3(gemmBlocks), 800, ldsBytes, stream>>>(S1bf, W2bf, Cbuf);
  k_lif<<<blocks1d((size_t)Bt * H), 256, 0, stream>>>(
      Cbuf, T, Bt, H, H, b2, v2, be2, s2r, S2bf, KP, nullptr);

  // 4) layer 3: cur3 = spk2 @ W3^T  (N padded 10 -> 32), direct kernel
  {
    const int nTN = NP3 / 32, tot = (int)(M / 64) * nTN; // 400 tiles
    k_gemm_bf16<<<dim3((tot + 7) / 8), 256, 0, stream>>>(S2bf, W3bf, Cbuf, KP, NP3, nTN, tot);
  }
  k_lif<<<blocks1d((size_t)Bt * OUT), 256, 0, stream>>>(
      Cbuf, T, Bt, OUT, NP3, b3, nullptr, be3, s3r, nullptr, 0, s3sum);
}
